// KimiMoEBlock_11227044511760
// MI455X (gfx1250) — compile-verified, matrix operand order checked by reading, not run on hardware
//
#include <hip/hip_runtime.h>
#include <hip/hip_bf16.h>
#include <cstdint>

#define H_DIM  2048
#define E_NUM  16
#define TOPK   4
#define I_DIM  1408
#define SHI    (2 * I_DIM)
#define T_TOK  1024

// Tiling
#define M_TILE 128
#define N_TILE 64
#define K_TILE 32
#define LDA    40   // padded LDS stride (bf16 elems): 80B rows keep 16B alignment
#define LDB    40

typedef __attribute__((ext_vector_type(16))) __bf16 v16bf;
typedef __attribute__((ext_vector_type(8)))  __bf16 v8bf;
typedef __attribute__((ext_vector_type(8)))  float  v8f;

__device__ __forceinline__ v8f wmma_bf16(v16bf a, v16bf b, v8f c) {
  // v_wmma_f32_16x16x32_bf16  D = A*B + C   (wave32, f32 accumulate)
  return __builtin_amdgcn_wmma_f32_16x16x32_bf16(
      /*neg_a=*/false, a, /*neg_b=*/false, b,
      /*c_mod=*/(short)0, c, /*reuse_a=*/false, /*reuse_b=*/false);
}

__device__ __forceinline__ v16bf load_frag_at(const __bf16* p) {
  union { v16bf v; v8bf h[2]; } u;
  u.h[0] = *(const v8bf*)(p);        // K 0..7  (lanes 0-15)  / 8..15  (16-31)
  u.h[1] = *(const v8bf*)(p + 16);   // K 16..23               / 24..31
  return u.v;
}

__device__ __forceinline__ int frag_off(int row_off, int ld) {
  int lane = threadIdx.x & 31;
  return (row_off + (lane & 15)) * ld + (lane >> 4) * 8;
}

// --------- software-pipeline staging helpers (regs <-> global/LDS) ---------
// All global indices are 32-bit off a uniform base pointer so the compiler can
// emit saddr+voffset (scale_offset) addressing instead of per-lane 64-bit adds.
struct ATile { uint2  v[4]; };   // 128x32 bf16 tile, qword granularity
struct WTile { float4 v[2]; };   // 32x64 fp32 tile, float4 granularity

// A: a_r = tid>>3 (0..31, +32/iter), a_c4 = tid&7 (qword column)
__device__ __forceinline__ void a_load(ATile& t, const __bf16* src,
                                       uint32_t row0, uint32_t col0,
                                       uint32_t ld, uint32_t a_r, uint32_t a_c4) {
#pragma unroll
  for (int i = 0; i < 4; ++i)
    t.v[i] = *(const uint2*)(src + (row0 + a_r + i * 32) * ld + col0 + a_c4 * 4);
}
__device__ __forceinline__ void a_store(__bf16* dst, const ATile& t,
                                        uint32_t a_r, uint32_t a_c4) {
#pragma unroll
  for (int i = 0; i < 4; ++i)
    *(uint2*)(dst + (a_r + i * 32) * LDA + a_c4 * 4) = t.v[i];
}
// W: w_k = tid>>4 (0..15, +16/iter), w_n4 = (tid&15)*4
__device__ __forceinline__ void w_load(WTile& t, const float* src,
                                       uint32_t k0, uint32_t n0, uint32_t ldw,
                                       uint32_t w_k, uint32_t w_n4) {
#pragma unroll
  for (int i = 0; i < 2; ++i)
    t.v[i] = *(const float4*)(src + (k0 + w_k + i * 16) * ldw + n0 + w_n4);
}
__device__ __forceinline__ void w_store(__bf16* dst, const WTile& t,
                                        uint32_t w_k, uint32_t w_n4) {
#pragma unroll
  for (int i = 0; i < 2; ++i) {
    uint32_t k = w_k + i * 16;
    dst[(w_n4 + 0) * LDB + k] = (__bf16)t.v[i].x;   // transposed [n][k]
    dst[(w_n4 + 1) * LDB + k] = (__bf16)t.v[i].y;
    dst[(w_n4 + 2) * LDB + k] = (__bf16)t.v[i].z;
    dst[(w_n4 + 3) * LDB + k] = (__bf16)t.v[i].w;
  }
}

// ---------------------------------------------------------------------------
// Router: logits -> fp32 softmax -> top-4 -> dense routing weights [T,E]
// ---------------------------------------------------------------------------
__global__ __launch_bounds__(256) void router_kernel(
    const float* __restrict__ x, const float* __restrict__ w_gate,
    float* __restrict__ rw) {
  int t = blockIdx.x * blockDim.x + threadIdx.x;
  if (t >= T_TOK) return;
  float logit[E_NUM];
#pragma unroll
  for (int e = 0; e < E_NUM; ++e) logit[e] = 0.f;
  for (int h = 0; h < H_DIM; ++h) {
    float xv = x[(size_t)t * H_DIM + h];
#pragma unroll
    for (int e = 0; e < E_NUM; ++e) logit[e] += xv * w_gate[(size_t)e * H_DIM + h];
  }
  float m = logit[0];
#pragma unroll
  for (int e = 1; e < E_NUM; ++e) m = fmaxf(m, logit[e]);
  float p[E_NUM], s = 0.f;
#pragma unroll
  for (int e = 0; e < E_NUM; ++e) { p[e] = __expf(logit[e] - m); s += p[e]; }
  float inv = 1.f / s;
#pragma unroll
  for (int e = 0; e < E_NUM; ++e) p[e] *= inv;
  float out[E_NUM];
#pragma unroll
  for (int e = 0; e < E_NUM; ++e) out[e] = 0.f;
#pragma unroll
  for (int k = 0; k < TOPK; ++k) {
    int best = 0; float bv = -1.f;
#pragma unroll
    for (int e = 0; e < E_NUM; ++e)
      if (p[e] > bv) { bv = p[e]; best = e; }
    out[best] = bv;
    p[best] = -1.f;
  }
#pragma unroll
  for (int e = 0; e < E_NUM; ++e) rw[(size_t)t * E_NUM + e] = out[e];
}

__global__ __launch_bounds__(256) void cvt_f32_bf16(
    const float* __restrict__ src, __bf16* __restrict__ dst, int n) {
  int i = blockIdx.x * blockDim.x + threadIdx.x;
  if (i < n) dst[i] = (__bf16)src[i];
}

// ---------------------------------------------------------------------------
// Fused gate/up GEMM + SwiGLU (+ routing-weight scale) -> bf16 activations.
// Grid: (N/64, n_experts). Weights [H x N] row-major fp32, converted on load.
// Double-buffered LDS pipeline: next-tile global loads issue before current
// tile's WMMA, LDS stores land in the alternate buffer, one barrier per step.
// ---------------------------------------------------------------------------
template <bool USE_RW>
__global__ __launch_bounds__(256) void gateup_kernel(
    const __bf16* __restrict__ xb, const float* __restrict__ Wg,
    const float* __restrict__ Wu, const float* __restrict__ rw,
    __bf16* __restrict__ Act, int N) {
  __shared__ __bf16 Xs[2][M_TILE * LDA];
  __shared__ __bf16 Bgs[2][N_TILE * LDB];
  __shared__ __bf16 Bus[2][N_TILE * LDB];

  const int e  = blockIdx.y;
  const int n0 = blockIdx.x * N_TILE;
  const float* wg = Wg + (size_t)e * H_DIM * N;
  const float* wu = Wu + (size_t)e * H_DIM * N;
  __bf16* act = Act + (size_t)e * T_TOK * N;

  const int tid  = threadIdx.x;
  const int wave = tid >> 5;
  const int mw = wave & 3, nw = wave >> 2;
  const int lane = tid & 31;
  const int nin  = lane & 15;
  const int mb   = (lane >> 4) * 8;

  const int aoff0 = frag_off(mw * 32 + 0,  LDA);
  const int aoff1 = frag_off(mw * 32 + 16, LDA);
  const int boff0 = frag_off(nw * 32 + 0,  LDB);
  const int boff1 = frag_off(nw * 32 + 16, LDB);

  const uint32_t a_r  = tid >> 3, a_c4 = tid & 7;
  const uint32_t w_k  = tid >> 4, w_n4 = (tid & 15) * 4;

  constexpr int KC = H_DIM / K_TILE;

  for (int mt = 0; mt < T_TOK / M_TILE; ++mt) {
    const uint32_t row0 = mt * M_TILE;
    v8f accg[2][2] = {};
    v8f accu[2][2] = {};

    ATile ar; WTile gr, ur;
    a_load(ar, xb, row0, 0, H_DIM, a_r, a_c4);
    w_load(gr, wg, 0, n0, N, w_k, w_n4);
    w_load(ur, wu, 0, n0, N, w_k, w_n4);
    a_store(Xs[0], ar, a_r, a_c4);
    w_store(Bgs[0], gr, w_k, w_n4);
    w_store(Bus[0], ur, w_k, w_n4);
    __syncthreads();

    for (int kc = 0; kc < KC; ++kc) {
      const int cur = kc & 1, nxt = cur ^ 1;
      const bool more = (kc + 1) < KC;
      if (more) {               // issue next-tile global loads early
        a_load(ar, xb, row0, (kc + 1) * K_TILE, H_DIM, a_r, a_c4);
        w_load(gr, wg, (kc + 1) * K_TILE, n0, N, w_k, w_n4);
        w_load(ur, wu, (kc + 1) * K_TILE, n0, N, w_k, w_n4);
      }
      // compute current tile (overlaps with in-flight VMEM)
      v16bf a0 = load_frag_at(Xs[cur] + aoff0);
      v16bf a1 = load_frag_at(Xs[cur] + aoff1);
      {
        v16bf b0 = load_frag_at(Bgs[cur] + boff0);
        v16bf b1 = load_frag_at(Bgs[cur] + boff1);
        accg[0][0] = wmma_bf16(a0, b0, accg[0][0]);
        accg[1][0] = wmma_bf16(a1, b0, accg[1][0]);
        accg[0][1] = wmma_bf16(a0, b1, accg[0][1]);
        accg[1][1] = wmma_bf16(a1, b1, accg[1][1]);
      }
      {
        v16bf b0 = load_frag_at(Bus[cur] + boff0);
        v16bf b1 = load_frag_at(Bus[cur] + boff1);
        accu[0][0] = wmma_bf16(a0, b0, accu[0][0]);
        accu[1][0] = wmma_bf16(a1, b0, accu[1][0]);
        accu[0][1] = wmma_bf16(a0, b1, accu[0][1]);
        accu[1][1] = wmma_bf16(a1, b1, accu[1][1]);
      }
      if (more) {
        a_store(Xs[nxt], ar, a_r, a_c4);
        w_store(Bgs[nxt], gr, w_k, w_n4);
        w_store(Bus[nxt], ur, w_k, w_n4);
      }
      __syncthreads();
    }

    // epilogue: silu(g)*u (*rw) -> bf16 ; fast rcp instead of IEEE divide
#pragma unroll
    for (int sm = 0; sm < 2; ++sm)
#pragma unroll
      for (int sn = 0; sn < 2; ++sn)
#pragma unroll
        for (int j = 0; j < 8; ++j) {
          int row = row0 + mw * 32 + sm * 16 + mb + j;     // token
          int col = n0 + nw * 32 + sn * 16 + nin;          // intermediate idx
          float g = accg[sm][sn][j];
          float u = accu[sm][sn][j];
          float a = g * __builtin_amdgcn_rcpf(1.f + __expf(-g)) * u;
          if constexpr (USE_RW) a *= rw[(size_t)row * E_NUM + e];
          act[(size_t)row * N + col] = (__bf16)a;
        }
  }
}

// ---------------------------------------------------------------------------
// Down projection: out[T,H] = sum_e ActR_e @ Wd_e  +  ActS @ Sd  (fp32 out)
// Grid: (H/64). Accumulates all 17 segments in registers, single store.
// ---------------------------------------------------------------------------
__global__ __launch_bounds__(256) void down_kernel(
    const __bf16* __restrict__ actR, const __bf16* __restrict__ actS,
    const float* __restrict__ wd, const float* __restrict__ sd,
    float* __restrict__ out) {
  __shared__ __bf16 As[2][M_TILE * LDA];
  __shared__ __bf16 Bs[2][N_TILE * LDB];

  const int n0 = blockIdx.x * N_TILE;
  const int tid  = threadIdx.x;
  const int wave = tid >> 5;
  const int mw = wave & 3, nw = wave >> 2;
  const int lane = tid & 31;
  const int nin  = lane & 15;
  const int mb   = (lane >> 4) * 8;

  const int aoff0 = frag_off(mw * 32 + 0,  LDA);
  const int aoff1 = frag_off(mw * 32 + 16, LDA);
  const int boff0 = frag_off(nw * 32 + 0,  LDB);
  const int boff1 = frag_off(nw * 32 + 16, LDB);

  const uint32_t a_r = tid >> 3, a_c4 = tid & 7;
  const uint32_t w_k = tid >> 4, w_n4 = (tid & 15) * 4;

  for (int mt = 0; mt < T_TOK / M_TILE; ++mt) {
    const uint32_t row0 = mt * M_TILE;
    v8f acc[2][2] = {};
    for (int seg = 0; seg <= E_NUM; ++seg) {
      const bool sh = (seg == E_NUM);
      const __bf16* A = sh ? actS : actR + (size_t)seg * T_TOK * I_DIM;
      const float*  W = sh ? sd   : wd   + (size_t)seg * I_DIM * H_DIM;
      const uint32_t ldA = sh ? SHI : I_DIM;
      const int KC       = ldA / K_TILE;

      ATile ar; WTile wr;
      a_load(ar, A, row0, 0, ldA, a_r, a_c4);
      w_load(wr, W, 0, n0, H_DIM, w_k, w_n4);
      a_store(As[0], ar, a_r, a_c4);
      w_store(Bs[0], wr, w_k, w_n4);
      __syncthreads();

      for (int kc = 0; kc < KC; ++kc) {
        const int cur = kc & 1, nxt = cur ^ 1;
        const bool more = (kc + 1) < KC;
        if (more) {
          a_load(ar, A, row0, (kc + 1) * K_TILE, ldA, a_r, a_c4);
          w_load(wr, W, (kc + 1) * K_TILE, n0, H_DIM, w_k, w_n4);
        }
        v16bf a0 = load_frag_at(As[cur] + aoff0);
        v16bf a1 = load_frag_at(As[cur] + aoff1);
        v16bf b0 = load_frag_at(Bs[cur] + boff0);
        v16bf b1 = load_frag_at(Bs[cur] + boff1);
        acc[0][0] = wmma_bf16(a0, b0, acc[0][0]);
        acc[1][0] = wmma_bf16(a1, b0, acc[1][0]);
        acc[0][1] = wmma_bf16(a0, b1, acc[0][1]);
        acc[1][1] = wmma_bf16(a1, b1, acc[1][1]);
        if (more) {
          a_store(As[nxt], ar, a_r, a_c4);
          w_store(Bs[nxt], wr, w_k, w_n4);
        }
        __syncthreads();
      }
    }
#pragma unroll
    for (int sm = 0; sm < 2; ++sm)
#pragma unroll
      for (int sn = 0; sn < 2; ++sn)
#pragma unroll
        for (int j = 0; j < 8; ++j) {
          int row = row0 + mw * 32 + sm * 16 + mb + j;
          int col = n0 + nw * 32 + sn * 16 + nin;
          out[(size_t)row * H_DIM + col] = acc[sm][sn][j];
        }
  }
}

// ---------------------------------------------------------------------------
extern "C" void kernel_launch(void* const* d_in, const int* in_sizes, int n_in,
                              void* d_out, int out_size, void* d_ws, size_t ws_size,
                              hipStream_t stream) {
  const float* x      = (const float*)d_in[0];   // [T,H]
  const float* w_gate = (const float*)d_in[1];   // [E,H]
  const float* wg     = (const float*)d_in[2];   // [E,H,I]
  const float* wu     = (const float*)d_in[3];   // [E,H,I]
  const float* wd     = (const float*)d_in[4];   // [E,I,H]
  const float* sg     = (const float*)d_in[5];   // [H,SHI]
  const float* su     = (const float*)d_in[6];   // [H,SHI]
  const float* sd     = (const float*)d_in[7];   // [SHI,H]
  float* out = (float*)d_out;                    // [T,H] fp32

  // workspace layout (all 16B-aligned sizes)
  char* ws = (char*)d_ws;
  __bf16* xb   = (__bf16*)ws;  ws += (size_t)T_TOK * H_DIM * 2;         // 4 MB
  float*  rw   = (float*)ws;   ws += (size_t)T_TOK * E_NUM * 4;         // 64 KB
  __bf16* actR = (__bf16*)ws;  ws += (size_t)E_NUM * T_TOK * I_DIM * 2; // 44 MB
  __bf16* actS = (__bf16*)ws;                                            // 5.5 MB

  // 1. router
  router_kernel<<<T_TOK / 256, 256, 0, stream>>>(x, w_gate, rw);

  // 2. x -> bf16
  {
    int n = T_TOK * H_DIM;
    cvt_f32_bf16<<<(n + 255) / 256, 256, 0, stream>>>(x, xb, n);
  }

  // 3. routed experts gate/up + SwiGLU + routing scale
  gateup_kernel<true><<<dim3(I_DIM / N_TILE, E_NUM), 256, 0, stream>>>(
      xb, wg, wu, rw, actR, I_DIM);

  // 4. shared experts gate/up + SwiGLU
  gateup_kernel<false><<<dim3(SHI / N_TILE, 1), 256, 0, stream>>>(
      xb, sg, su, rw, actS, SHI);

  // 5. down projections (routed sum + shared), fp32 store
  down_kernel<<<dim3(H_DIM / N_TILE), 256, 0, stream>>>(actR, actS, wd, sd, out);
}
